// JointProbabilisticLoss_61143154426441
// MI455X (gfx1250) — compile-verified, matrix operand order checked by reading, not run on hardware
//
#include <hip/hip_runtime.h>
#include <hip/hip_bf16.h>
#include <stdint.h>

typedef __attribute__((ext_vector_type(16))) _Float16 v16h;
typedef __attribute__((ext_vector_type(8)))  float    v8f;

#define N_    8
#define J_    17
#define D_    64
#define H_    64
#define W_    64
#define K_    (D_ * H_ * W_)        // 262144
#define S_    16                    // NUM_SAMPLES
#define ROWS_ (N_ * J_)             // 136
#define TPB   256
#define PER_ROUND (TPB * 16)        // 4096 elements per WMMA round per WG
#define ROUNDS (K_ / PER_ROUND)     // 64 (exact -> full EXEC at every WMMA)

__device__ __forceinline__ uint32_t mix32(uint32_t a) {
    a ^= a >> 16; a *= 0x7feb352du;
    a ^= a >> 15; a *= 0x846ca68bu;
    a ^= a >> 16;
    return a;
}

// ---------------------------------------------------------------------------
// Kernel 1: per-row fused  max -> (WMMA sum-exp  +  16 exponential races)
// One 256-thread workgroup (8 wave32) per (n,j) row.
// ---------------------------------------------------------------------------
__global__ __launch_bounds__(TPB)
void jpl_row_kernel(const float* __restrict__ preds,
                    float* __restrict__ row_max,
                    float* __restrict__ row_sumexp,
                    int*   __restrict__ samp_idx /* [ROWS_][S_] */)
{
    const int row = blockIdx.x;
    const float* __restrict__ x = preds + (size_t)row * K_;
    const int tid = threadIdx.x;

    __shared__ float sf[TPB];
    __shared__ int   si[TPB];
    __shared__ float s_rowmax;

    // ---- Phase 1: row max (streaming, coalesced) ----
    float m = -INFINITY;
    for (int i = tid; i < K_; i += TPB) m = fmaxf(m, x[i]);
    sf[tid] = m;
    __syncthreads();
    for (int off = TPB / 2; off > 0; off >>= 1) {
        if (tid < off) sf[tid] = fmaxf(sf[tid], sf[tid + off]);
        __syncthreads();
    }
    if (tid == 0) s_rowmax = sf[0];
    __syncthreads();
    const float rmax = s_rowmax;

    // ---- Phase 2: fused sum(exp(x-rmax)) via WMMA + 16 Gumbel/Exp races ----
    v16h ones;
#pragma unroll
    for (int i = 0; i < 16; ++i) ones[i] = (_Float16)1.0f;

    v8f acc = {};   // f32 accumulator: D[m][n] += sum_k A[m][k] (B = ones)

    float bestKey[S_];
    int   bestIdx[S_];
#pragma unroll
    for (int s = 0; s < S_; ++s) { bestKey[s] = INFINITY; bestIdx[s] = 0; }

    const uint32_t rowSeed = 0x9E3779B1u * (uint32_t)(row + 1);

    for (int r = 0; r < ROUNDS; ++r) {
        const int base = r * PER_ROUND;
        // prefetch next round's cacheline for this lane (speculative, OOB-safe)
        __builtin_prefetch((const void*)(x + base + PER_ROUND + tid), 0, 0);

        v16h a;
#pragma unroll
        for (int j = 0; j < 16; ++j) {
            const int k   = base + j * TPB + tid;    // coalesced across the wave
            const float v = x[k];
            const float e = __expf(v - rmax);        // shared by sum + all races
            a[j] = (_Float16)e;
            const float rinv = __builtin_amdgcn_rcpf(e);  // exp(rmax - v)
#pragma unroll
            for (int s = 0; s < S_; ++s) {
                // counter-based hash RNG: deterministic across graph replays
                uint32_t h = mix32((uint32_t)k ^ rowSeed ^ (0x85EBCA77u * (uint32_t)(s + 1)));
                float u   = (float)(h >> 9) * (1.0f / 8388608.0f) + 5.9604645e-8f; // (0,1]
                float g   = -__logf(u);               // Exp(1) variate
                float key = g * rinv;                 // Exp(1)/p_k race
                if (key < bestKey[s]) { bestKey[s] = key; bestIdx[s] = k; }
            }
        }
        // 512 f32 adds per instruction per wave; EXEC is all-1s (uniform loop)
        acc = __builtin_amdgcn_wmma_f32_16x16x32_f16(
                  false, a, false, ones, (short)0, acc, false, false);
    }

    // collapse WMMA accumulator: every column of D replicates the row sums,
    // so sum over all lanes/VGPRs == 16 * (total of all A elements fed).
    float s8 = acc[0] + acc[1] + acc[2] + acc[3] + acc[4] + acc[5] + acc[6] + acc[7];
    sf[tid] = s8;
    __syncthreads();
    for (int off = TPB / 2; off > 0; off >>= 1) {
        if (tid < off) sf[tid] += sf[tid + off];
        __syncthreads();
    }
    if (tid == 0) {
        row_max[row]    = rmax;
        row_sumexp[row] = sf[0] * (1.0f / 16.0f);
    }

    // per-sample argmin across the workgroup
    for (int s = 0; s < S_; ++s) {
        __syncthreads();
        sf[tid] = bestKey[s];
        si[tid] = bestIdx[s];
        __syncthreads();
        for (int off = TPB / 2; off > 0; off >>= 1) {
            if (tid < off && sf[tid + off] < sf[tid]) {
                sf[tid] = sf[tid + off];
                si[tid] = si[tid + off];
            }
            __syncthreads();
        }
        if (tid == 0) samp_idx[row * S_ + s] = si[0];
    }
}

// ---------------------------------------------------------------------------
// Kernel 2: tiny finalize (one workgroup). Reproduces reference reduction:
//   L[s,n]   = sum_j sum_c |coord - gt| * vis
//   logp     = x[idx] - (rowmax + log(sumexp))
//   out      = mean over (s,n,j) of L[s,n] / (-logp[s,n,j])
// ---------------------------------------------------------------------------
__global__ __launch_bounds__(TPB)
void jpl_finalize_kernel(const float* __restrict__ preds,
                         const float* __restrict__ gt,   // (ROWS_,3)
                         const float* __restrict__ vis,  // (ROWS_,3)
                         const float* __restrict__ row_max,
                         const float* __restrict__ row_sumexp,
                         const int*   __restrict__ samp_idx,
                         float* __restrict__ out)
{
    __shared__ float s_abs [S_ * ROWS_];   // 2176
    __shared__ float s_logp[S_ * ROWS_];   // 2176
    __shared__ float s_L   [S_ * N_];      // 128
    __shared__ float s_red [TPB];

    const int tid = threadIdx.x;

    for (int t = tid; t < S_ * ROWS_; t += TPB) {
        const int s   = t / ROWS_;
        const int row = t % ROWS_;
        const int idx = samp_idx[row * S_ + s];
        const float xv   = preds[(size_t)row * K_ + idx];
        const float logZ = row_max[row] + __logf(row_sumexp[row]);
        s_logp[t] = xv - logZ;

        const float cx = (float)(idx % W_)          / (float)W_ - 0.5f;
        const float cy = (float)((idx / W_) % H_)   / (float)H_ - 0.5f;
        const float cz = (float)(idx / (W_ * H_))   / (float)D_ - 0.5f;
        const float* g = gt  + row * 3;
        const float* v = vis + row * 3;
        s_abs[t] = fabsf(cx - g[0]) * v[0]
                 + fabsf(cy - g[1]) * v[1]
                 + fabsf(cz - g[2]) * v[2];
    }
    __syncthreads();

    for (int t = tid; t < S_ * N_; t += TPB) {
        const int s = t / N_;
        const int n = t % N_;
        float L = 0.0f;
        for (int j = 0; j < J_; ++j) L += s_abs[s * ROWS_ + n * J_ + j];
        s_L[t] = L;
    }
    __syncthreads();

    float accv = 0.0f;
    for (int t = tid; t < S_ * ROWS_; t += TPB) {
        const int s   = t / ROWS_;
        const int row = t % ROWS_;
        const int n   = row / J_;
        accv += s_L[s * N_ + n] / (-s_logp[t]);
    }
    s_red[tid] = accv;
    __syncthreads();
    for (int off = TPB / 2; off > 0; off >>= 1) {
        if (tid < off) s_red[tid] += s_red[tid + off];
        __syncthreads();
    }
    if (tid == 0) out[0] = s_red[0] / (float)(S_ * ROWS_);
}

// ---------------------------------------------------------------------------
extern "C" void kernel_launch(void* const* d_in, const int* in_sizes, int n_in,
                              void* d_out, int out_size, void* d_ws, size_t ws_size,
                              hipStream_t stream)
{
    const float* preds = (const float*)d_in[0];
    const float* gt    = (const float*)d_in[1];
    const float* vis   = (const float*)d_in[2];

    // workspace layout: [ROWS_] max | [ROWS_] sumexp | [ROWS_*S_] indices
    float* rmax = (float*)d_ws;
    float* rsum = rmax + ROWS_;
    int*   sidx = (int*)(rsum + ROWS_);

    jpl_row_kernel<<<ROWS_, TPB, 0, stream>>>(preds, rmax, rsum, sidx);
    jpl_finalize_kernel<<<1, TPB, 0, stream>>>(preds, gt, vis, rmax, rsum, sidx,
                                               (float*)d_out);
}
